// MambaBlock3D_71614284694004
// MI455X (gfx1250) — matrix operand order repaired and checked by reading, not compiled, not run on hardware
//
#include <hip/hip_runtime.h>

typedef __attribute__((ext_vector_type(16))) _Float16 v16h;
typedef __attribute__((ext_vector_type(8)))  _Float16 v8h;
typedef __attribute__((ext_vector_type(8)))  float    v8f;

#define LTOK 13824   // 24*24*24 tokens
#define CCH  128     // channels
#define DI   256     // d_inner
#define DSN  16      // d_state
#define NXP  48      // x_proj rows padded 40 -> 48 (3 N-tiles of 16)
#define CHL  64      // scan chunk length
#define NCH  (LTOK / CHL)   // 216 chunks

// ---------------- LayerNorm over C, emit f16 tokens (L x 128) ----------------
__global__ void __launch_bounds__(256) ln_kernel(const float* __restrict__ x,
                                                 const float* __restrict__ lnw,
                                                 const float* __restrict__ lnb,
                                                 _Float16* __restrict__ tnh) {
  int l = blockIdx.x * blockDim.x + threadIdx.x;
  if (l >= LTOK) return;
  float s = 0.f, s2 = 0.f;
  for (int c = 0; c < CCH; ++c) {
    float v = x[(size_t)c * LTOK + l];
    s += v; s2 += v * v;
  }
  float mu  = s * (1.f / CCH);
  float var = s2 * (1.f / CCH) - mu * mu;
  float inv = rsqrtf(var + 1e-5f);
  for (int c = 0; c < CCH; ++c) {
    float v = x[(size_t)c * LTOK + l];
    tnh[(size_t)l * CCH + c] = (_Float16)(((v - mu) * inv) * lnw[c] + lnb[c]);
  }
}

// ---------------- Weight conversion f32 -> f16 (x_proj zero-padded to 48) ----
__global__ void __launch_bounds__(256) convert_w_kernel(const float* __restrict__ w1,
                                                        const float* __restrict__ w2,
                                                        const float* __restrict__ w3,
                                                        _Float16* __restrict__ w1h,
                                                        _Float16* __restrict__ w2h,
                                                        _Float16* __restrict__ w3h) {
  int i = blockIdx.x * blockDim.x + threadIdx.x;
  if (i < 512 * 128) w1h[i] = (_Float16)w1[i];
  if (i < NXP * DI)  w2h[i] = (i < 40 * DI) ? (_Float16)w2[i] : (_Float16)0.f;
  if (i < CCH * DI)  w3h[i] = (_Float16)w3[i];
}

// ---------------- WMMA GEMM: D(M x N) = A(M x K) * W(N x K)^T ----------------
// Each wave computes a 16 x (16*NT) strip, reusing the A fragment across NT
// B tiles.  MODE 0: f16 store, 1: f32 store,
// MODE 2: transposed store with residual add (out[n*ldt+m] = resid + acc).
template<int K, int NT, int MODE>
__global__ void __launch_bounds__(256) gemm_kernel(
    const _Float16* __restrict__ A, int lda,
    const _Float16* __restrict__ W, int ldb,
    int Mt, int Ng,
    _Float16* __restrict__ Oh, float* __restrict__ Of, int ldo,
    const float* __restrict__ resid, float* __restrict__ outT, int ldt)
{
  int wave = blockIdx.x * (blockDim.x >> 5) + (threadIdx.x >> 5);
  if (wave >= Mt * Ng) return;
  int tm = wave % Mt;
  int tg = wave / Mt;
  int lane = threadIdx.x & 31;
  int hi16 = lane >> 4;     // which half of the wave
  int r    = lane & 15;     // row (A) / col (B, D)

  // A fragment: lane r holds row m=r; the half selects K = 16*hi16..+15,
  // contiguous along K (same order as B).
  const _Float16* arow = A + (size_t)(tm * 16 + r) * lda + hi16 * 16;
  // B fragment: lane r holds column n=r; hi16 selects K 0..15 / 16..31,
  // contiguous along K because we multiply by W^T with W row-major (N x K).
  const _Float16* brow[NT];
#pragma unroll
  for (int j = 0; j < NT; ++j)
    brow[j] = W + (size_t)((tg * NT + j) * 16 + r) * ldb + hi16 * 16;

  v8f acc[NT];
#pragma unroll
  for (int j = 0; j < NT; ++j) acc[j] = v8f{};

#pragma unroll
  for (int k = 0; k < K; k += 32) {
    v8h alo = *(const v8h*)(arow + k);
    v8h ahi = *(const v8h*)(arow + k + 8);
    v16h af;
#pragma unroll
    for (int i = 0; i < 8; ++i) { af[i] = alo[i]; af[i + 8] = ahi[i]; }
#pragma unroll
    for (int j = 0; j < NT; ++j) {
      v16h bf = *(const v16h*)(brow[j] + k);
      acc[j] = __builtin_amdgcn_wmma_f32_16x16x32_f16(false, af, false, bf,
                                                      (short)0, acc[j], false, false);
    }
  }

#pragma unroll
  for (int j = 0; j < NT; ++j) {
    int gn = (tg * NT + j) * 16 + r;
#pragma unroll
    for (int v = 0; v < 8; ++v) {
      int gm = tm * 16 + v + 8 * hi16;   // C/D layout: VGPR v holds M=v / 8+v
      float val = acc[j][v];
      if (MODE == 0)      Oh[(size_t)gm * ldo + gn] = (_Float16)val;
      else if (MODE == 1) Of[(size_t)gm * ldo + gn] = val;
      else                outT[(size_t)gn * ldt + gm] =
                              resid[(size_t)gn * ldt + gm] + val;
    }
  }
}

// ---------------- Depthwise causal conv (4 taps) + SiLU ----------------------
__global__ void __launch_bounds__(256) conv_silu_kernel(const _Float16* __restrict__ xz,
                                                        const float* __restrict__ cw,
                                                        const float* __restrict__ cb,
                                                        float* __restrict__ xaf,
                                                        _Float16* __restrict__ xah) {
  int i = blockIdx.x * blockDim.x + threadIdx.x;
  if (i >= LTOK * DI) return;
  int d = i & (DI - 1);
  int l = i >> 8;
  float acc = cb[d];
#pragma unroll
  for (int k = 0; k < 4; ++k) {
    int ls = l - 3 + k;
    if (ls >= 0) acc += (float)xz[(size_t)ls * 512 + d] * cw[d * 4 + k];
  }
  float sa = acc / (1.f + __expf(-acc));   // silu
  xaf[i] = sa;
  xah[i] = (_Float16)sa;
}

// ---------------- dt = softplus(dt_in @ dt_proj^T + b)  (K = 8, VALU) --------
__global__ void __launch_bounds__(256) dt_kernel(const float* __restrict__ xdbl,
                                                 const float* __restrict__ dw,
                                                 const float* __restrict__ db,
                                                 float* __restrict__ dt) {
  int i = blockIdx.x * blockDim.x + threadIdx.x;
  if (i >= LTOK * DI) return;
  int d = i & (DI - 1);
  int l = i >> 8;
  float acc = db[d];
#pragma unroll
  for (int r = 0; r < 8; ++r) acc += xdbl[(size_t)l * NXP + r] * dw[d * 8 + r];
  dt[i] = (acc > 20.f) ? acc : log1pf(__expf(acc));
}

// ======================= Chunked parallel selective scan =====================
// Linear recurrence h_l = a_l h_{l-1} + b_l with a_l = exp(dt*A) in (0,1).
// Pass 1 (NCH-way parallel): per chunk, coefficient product P and zero-start
// partial state HP.  Pass 2 (sequential over 216 chunks): chunk start states
// HS.  Pass 3 (NCH-way parallel): replay with correct start state, emit y.

__global__ void __launch_bounds__(256) scan_pass1(const float* __restrict__ xdbl,
                                                  const float* __restrict__ dtp,
                                                  const float* __restrict__ xaf,
                                                  const float* __restrict__ A_log,
                                                  float* __restrict__ P,
                                                  float* __restrict__ HP) {
  __shared__ float sB[2][16];
  int c = blockIdx.x;
  int d = threadIdx.x;
  int l0 = c * CHL;
  float Ad[DSN], h[DSN], p[DSN];
#pragma unroll
  for (int s = 0; s < DSN; ++s) {
    Ad[s] = -__expf(A_log[d * DSN + s]);
    h[s] = 0.f; p[s] = 1.f;
  }
  if (d < 16) sB[0][d] = xdbl[(size_t)l0 * NXP + 8 + d];

  for (int t = 0; t < CHL; ++t) {
    int l = l0 + t;
    __syncthreads();
    int pb = t & 1;
    float Bl[DSN];
#pragma unroll
    for (int s = 0; s < DSN; ++s) Bl[s] = sB[pb][s];
    if (d < 16 && (t + 1) < CHL)
      sB[1 - pb][d] = xdbl[(size_t)(l + 1) * NXP + 8 + d];
    if (t + 8 < CHL) {
      __builtin_prefetch(dtp + (size_t)(l + 8) * DI + d, 0, 0);
      __builtin_prefetch(xaf + (size_t)(l + 8) * DI + d, 0, 0);
    }
    float dtd = dtp[(size_t)l * DI + d];
    float xad = xaf[(size_t)l * DI + d];
    float u = dtd * xad;
#pragma unroll
    for (int s = 0; s < DSN; ++s) {
      float a = __expf(dtd * Ad[s]);
      h[s] = a * h[s] + u * Bl[s];
      p[s] *= a;
    }
  }
  size_t base = ((size_t)c * DI + d) * DSN;
#pragma unroll
  for (int s = 0; s < DSN; ++s) { P[base + s] = p[s]; HP[base + s] = h[s]; }
}

__global__ void __launch_bounds__(256, 1) scan_pass2(const float* __restrict__ P,
                                                     const float* __restrict__ HP,
                                                     float* __restrict__ HS) {
  int d = threadIdx.x;
  float h[DSN];
#pragma unroll
  for (int s = 0; s < DSN; ++s) h[s] = 0.f;
  for (int c = 0; c < NCH; ++c) {
    size_t base = ((size_t)c * DI + d) * DSN;
#pragma unroll
    for (int s = 0; s < DSN; ++s) HS[base + s] = h[s];
#pragma unroll
    for (int s = 0; s < DSN; ++s) h[s] = P[base + s] * h[s] + HP[base + s];
  }
}

__global__ void __launch_bounds__(256) scan_pass3(const float* __restrict__ xdbl,
                                                  const float* __restrict__ dtp,
                                                  const float* __restrict__ xaf,
                                                  const _Float16* __restrict__ xz,
                                                  const float* __restrict__ A_log,
                                                  const float* __restrict__ Dp,
                                                  const float* __restrict__ HS,
                                                  _Float16* __restrict__ yh) {
  __shared__ float sBC[2][32];   // B (0..15) and C (16..31) for one token
  int c = blockIdx.x;
  int d = threadIdx.x;
  int l0 = c * CHL;
  size_t base = ((size_t)c * DI + d) * DSN;
  float Ad[DSN], h[DSN];
#pragma unroll
  for (int s = 0; s < DSN; ++s) {
    Ad[s] = -__expf(A_log[d * DSN + s]);
    h[s] = HS[base + s];
  }
  float Dd = Dp[d];
  if (d < 32) sBC[0][d] = xdbl[(size_t)l0 * NXP + 8 + d];  // cols 8..39 = B,C

  for (int t = 0; t < CHL; ++t) {
    int l = l0 + t;
    __syncthreads();
    int pb = t & 1;
    float Bl[DSN], Cl[DSN];
#pragma unroll
    for (int s = 0; s < DSN; ++s) { Bl[s] = sBC[pb][s]; Cl[s] = sBC[pb][16 + s]; }
    if (d < 32 && (t + 1) < CHL)
      sBC[1 - pb][d] = xdbl[(size_t)(l + 1) * NXP + 8 + d];
    if (t + 8 < CHL) {
      __builtin_prefetch(dtp + (size_t)(l + 8) * DI + d, 0, 0);
      __builtin_prefetch(xaf + (size_t)(l + 8) * DI + d, 0, 0);
    }
    float dtd = dtp[(size_t)l * DI + d];
    float xad = xaf[(size_t)l * DI + d];
    float u = dtd * xad;
    float y = 0.f;
#pragma unroll
    for (int s = 0; s < DSN; ++s) {
      h[s] = __expf(dtd * Ad[s]) * h[s] + u * Bl[s];
      y += h[s] * Cl[s];
    }
    y += xad * Dd;
    float zf = (float)xz[(size_t)l * 512 + DI + d];
    y *= zf / (1.f + __expf(-zf));          // * silu(z)
    yh[(size_t)l * DI + d] = (_Float16)y;
  }
}

extern "C" void kernel_launch(void* const* d_in, const int* in_sizes, int n_in,
                              void* d_out, int out_size, void* d_ws, size_t ws_size,
                              hipStream_t stream) {
  (void)in_sizes; (void)n_in; (void)out_size; (void)ws_size;
  const float* x      = (const float*)d_in[0];
  const float* ln_w   = (const float*)d_in[1];
  const float* ln_b   = (const float*)d_in[2];
  const float* w_in   = (const float*)d_in[3];   // (512,128)
  const float* conv_w = (const float*)d_in[4];   // (256,4)
  const float* conv_b = (const float*)d_in[5];
  const float* w_xp   = (const float*)d_in[6];   // (40,256)
  const float* w_dt   = (const float*)d_in[7];   // (256,8)
  const float* b_dt   = (const float*)d_in[8];
  const float* A_log  = (const float*)d_in[9];   // (256,16)
  const float* D_par  = (const float*)d_in[10];
  const float* w_out  = (const float*)d_in[11];  // (128,256)
  float* outp = (float*)d_out;

  char* ws = (char*)d_ws;
  size_t off = 0;
  auto alloc = [&](size_t bytes) {
    size_t o = off; off += (bytes + 255) & ~(size_t)255; return o;
  };
  _Float16* w1h  = (_Float16*)(ws + alloc((size_t)512 * 128 * 2));
  _Float16* w2h  = (_Float16*)(ws + alloc((size_t)NXP * DI * 2));
  _Float16* w3h  = (_Float16*)(ws + alloc((size_t)CCH * DI * 2));
  _Float16* tnh  = (_Float16*)(ws + alloc((size_t)LTOK * CCH * 2));
  _Float16* xzh  = (_Float16*)(ws + alloc((size_t)LTOK * 512 * 2));
  _Float16* xah  = (_Float16*)(ws + alloc((size_t)LTOK * DI * 2));
  float*    xaf  = (float*)   (ws + alloc((size_t)LTOK * DI * 4));
  float*    xdbl = (float*)   (ws + alloc((size_t)LTOK * NXP * 4));
  float*    dtb  = (float*)   (ws + alloc((size_t)LTOK * DI * 4));
  _Float16* yhb  = (_Float16*)(ws + alloc((size_t)LTOK * DI * 2));
  float*    Pb   = (float*)   (ws + alloc((size_t)NCH * DI * DSN * 4));
  float*    HPb  = (float*)   (ws + alloc((size_t)NCH * DI * DSN * 4));
  float*    HSb  = (float*)   (ws + alloc((size_t)NCH * DI * DSN * 4));

  const int Mt = LTOK / 16;  // 864 M-tiles

  convert_w_kernel<<<(512 * 128 + 255) / 256, 256, 0, stream>>>(
      w_in, w_xp, w_out, w1h, w2h, w3h);

  ln_kernel<<<(LTOK + 255) / 256, 256, 0, stream>>>(x, ln_w, ln_b, tnh);

  // in_proj: (L x 128) * (512 x 128)^T -> xz f16 (L x 512); 16x64 per wave
  gemm_kernel<128, 4, 0><<<(Mt * 8) / 8, 256, 0, stream>>>(
      tnh, 128, w1h, 128, Mt, 8, xzh, nullptr, 512, nullptr, nullptr, 0);

  conv_silu_kernel<<<(LTOK * DI) / 256, 256, 0, stream>>>(
      xzh, conv_w, conv_b, xaf, xah);

  // x_proj: (L x 256) * (48 x 256)^T -> x_dbl f32 (L x 48); 16x48 per wave
  gemm_kernel<256, 3, 1><<<(Mt + 7) / 8, 256, 0, stream>>>(
      xah, 256, w2h, 256, Mt, 1, nullptr, xdbl, NXP, nullptr, nullptr, 0);

  dt_kernel<<<(LTOK * DI) / 256, 256, 0, stream>>>(xdbl, w_dt, b_dt, dtb);

  scan_pass1<<<NCH, 256, 0, stream>>>(xdbl, dtb, xaf, A_log, Pb, HPb);
  scan_pass2<<<1, 256, 0, stream>>>(Pb, HPb, HSb);
  scan_pass3<<<NCH, 256, 0, stream>>>(xdbl, dtb, xaf, xzh, A_log, D_par,
                                      HSb, yhb);

  // out_proj: (L x 256) * (128 x 256)^T, transposed store + residual into d_out
  gemm_kernel<256, 4, 2><<<(Mt * 2) / 8, 256, 0, stream>>>(
      yhb, 256, w3h, 256, Mt, 2, nullptr, nullptr, 0, x, outp, LTOK);
}
